// UnifiedAttentionAggregator_88399016886473
// MI455X (gfx1250) — matrix operand order repaired and checked by reading, not compiled, not run on hardware
//
#include <hip/hip_runtime.h>

#define N_TOT 131072
#define NF    512
#define DQ    128
#define NBAG  32

typedef __attribute__((ext_vector_type(16))) __bf16 v16bf;
typedef __attribute__((ext_vector_type(8)))  float  v8f;
typedef int v4i __attribute__((__vector_size__(4 * sizeof(int))));

#define AS1 __attribute__((address_space(1)))
#define AS3 __attribute__((address_space(3)))

#if defined(__has_builtin)
#if __has_builtin(__builtin_amdgcn_global_load_async_to_lds_b128)
#define HAVE_ASYNC_LDS 1
#endif
#endif

// ---- monotonic float<->uint encoding for atomicMax on floats ----
__device__ __forceinline__ unsigned encf(float f) {
  unsigned u = __float_as_uint(f);
  return (u & 0x80000000u) ? ~u : (u | 0x80000000u);
}
__device__ __forceinline__ float decf(unsigned e) {
  return (e & 0x80000000u) ? __uint_as_float(e & 0x7FFFFFFFu)
                           : __uint_as_float(~e);
}

// ---------------- K0: offsets prefix + zero accumulators ----------------
__global__ void init_kernel(const int* __restrict__ split, int* offs,
                            unsigned* menc, float* ssum, float* pacc) {
  int t = threadIdx.x;
  if (t == 0) {
    int a = 0; offs[0] = 0;
    for (int i = 0; i < NBAG; ++i) { a += split[i]; offs[i + 1] = a; }
  }
  if (t < NBAG) { menc[t] = 0u; ssum[t] = 0.f; pacc[t] = 0.f; }
}

// ---------------- K1: c[n] = feats[n] . w_ins + b_ins (GEMV) ----------------
__global__ __launch_bounds__(256) void logits_kernel(
    const float* __restrict__ feats, const float* __restrict__ w_ins,
    const float* __restrict__ b_ins, float* __restrict__ cout) {
  int wave = threadIdx.x >> 5, lane = threadIdx.x & 31;
  int row = blockIdx.x * 8 + wave;
  const float* fr = feats + (size_t)row * NF;
  float acc = 0.f;
#pragma unroll
  for (int j = 0; j < 4; ++j) {
    int k = j * 128 + lane * 4;
    float4 x = *(const float4*)(fr + k);
    float4 w = *(const float4*)(w_ins + k);
    acc += x.x * w.x + x.y * w.y + x.z * w.z + x.w * w.w;
  }
#pragma unroll
  for (int m = 16; m; m >>= 1) acc += __shfl_xor(acc, m);
  if (lane == 0) cout[row] = acc + b_ins[0];
}

// ------------- K2: per-bag argmax of c (first-index tie-break) -------------
__global__ __launch_bounds__(256) void bag_argmax_kernel(
    const float* __restrict__ c, const int* __restrict__ offs, int* crit) {
  int b = blockIdx.x;
  int s = offs[b], e = offs[b + 1];
  float bv = -3.4e38f; int bi = N_TOT;
  for (int n = s + threadIdx.x; n < e; n += 256) {
    float v = c[n];
    if (v > bv || (v == bv && n < bi)) { bv = v; bi = n; }
  }
  __shared__ float sv[256];
  __shared__ int   si[256];
  sv[threadIdx.x] = bv; si[threadIdx.x] = bi;
  __syncthreads();
  for (int st = 128; st > 0; st >>= 1) {
    if (threadIdx.x < st) {
      float ov = sv[threadIdx.x + st]; int oi = si[threadIdx.x + st];
      if (ov > sv[threadIdx.x] ||
          (ov == sv[threadIdx.x] && oi < si[threadIdx.x])) {
        sv[threadIdx.x] = ov; si[threadIdx.x] = oi;
      }
    }
    __syncthreads();
  }
  if (threadIdx.x == 0) crit[b] = si[0];
}

// ---- K3: q_max[b] = feats[crit]@w_q + b_q ; r_b = w_q @ q_max ; beta_b ----
__global__ __launch_bounds__(256) void qmax_kernel(
    const float* __restrict__ feats, const float* __restrict__ w_q,
    const float* __restrict__ b_q, const int* __restrict__ crit,
    float* qmax, float* rvec, float* beta) {
  int b = blockIdx.x;
  int ci = crit[b];
  __shared__ float qm[DQ];
  if (threadIdx.x < DQ) {
    int d = threadIdx.x;
    float acc = b_q[d];
    const float* fr = feats + (size_t)ci * NF;
    for (int k = 0; k < NF; ++k) acc += fr[k] * w_q[k * DQ + d];
    qm[d] = acc; qmax[b * DQ + d] = acc;
  }
  __syncthreads();
  for (int k = threadIdx.x; k < NF; k += 256) {
    float acc = 0.f;
    for (int d = 0; d < DQ; ++d) acc += w_q[k * DQ + d] * qm[d];
    rvec[b * NF + k] = acc;
  }
  if (threadIdx.x == 0) {
    float acc = 0.f;
    for (int d = 0; d < DQ; ++d) acc += b_q[d] * qm[d];
    beta[b] = acc;
  }
}

// ---- K4: A[n] = (feats[n].r_b + beta_b)/sqrt(dq), track per-bag max ----
__global__ __launch_bounds__(256) void scores_kernel(
    const float* __restrict__ feats, const float* __restrict__ rvec,
    const float* __restrict__ beta, const int* __restrict__ offs,
    float* __restrict__ Abuf, unsigned* menc) {
  int wave = threadIdx.x >> 5, lane = threadIdx.x & 31;
  int row = blockIdx.x * 8 + wave;
  int b = 0;
  if (lane == 0) { while (offs[b + 1] <= row) b++; }
  b = __shfl(b, 0);
  const float* fr = feats + (size_t)row * NF;
  const float* rv = rvec + b * NF;
  float acc = 0.f;
#pragma unroll
  for (int j = 0; j < 4; ++j) {
    int k = j * 128 + lane * 4;
    float4 x = *(const float4*)(fr + k);
    float4 w = *(const float4*)(rv + k);
    acc += x.x * w.x + x.y * w.y + x.z * w.z + x.w * w.w;
  }
#pragma unroll
  for (int m = 16; m; m >>= 1) acc += __shfl_xor(acc, m);
  if (lane == 0) {
    float a = (acc + beta[b]) * 0.08838834764831845f;  // 1/sqrt(128)
    Abuf[row] = a;
    atomicMax(&menc[b], encf(a));
  }
}

// ---- K5: s_b = sum exp(A - m_b) ----
__global__ __launch_bounds__(256) void softsum_kernel(
    const float* __restrict__ Abuf, const int* __restrict__ offs,
    const unsigned* __restrict__ menc, float* ssum) {
  int n = blockIdx.x * 256 + threadIdx.x;
  int b = 0; while (offs[b + 1] <= n) b++;
  float e = __expf(Abuf[n] - decf(menc[b]));
  __shared__ float red[256];
  __shared__ int bs[2];
  if (threadIdx.x == 0)   bs[0] = b;
  if (threadIdx.x == 255) bs[1] = b;
  red[threadIdx.x] = e;
  __syncthreads();
  bool uniform = (bs[0] == bs[1]);
  for (int st = 128; st > 0; st >>= 1) {
    if (threadIdx.x < st) red[threadIdx.x] += red[threadIdx.x + st];
    __syncthreads();
  }
  if (uniform) { if (threadIdx.x == 0) atomicAdd(&ssum[b], red[0]); }
  else atomicAdd(&ssum[b], e);
}

// ---- K6: pack w_v (fp32 row-major [k][f]) into per-lane bf16 B-fragment layout ----
// Fragment layout (this code's contiguous k order, same as A; B 32x16): lane = (f%16) + 16*(kk>=16),
// element within v16bf = kk%16. Entry e = kb*32 + (f/16); 1 KB per entry.
__global__ __launch_bounds__(256) void wvpack_kernel(
    const float* __restrict__ w_v, unsigned short* __restrict__ wvpack) {
  int idx = blockIdx.x * 256 + threadIdx.x;      // over 512*512
  int k = idx >> 9, f = idx & 511;
  int kb = k >> 5, kk = k & 31, ft = f >> 4, fl = f & 15;
  int lane = fl + ((kk >= 16) ? 16 : 0);
  int elem = kk & 15;
  size_t dst = ((size_t)(kb * 32 + ft) * 32 + lane) * 16 + elem;
  __bf16 h = (__bf16)w_v[idx];
  unsigned short u;
  __builtin_memcpy(&u, &h, 2);
  wvpack[dst] = u;
}

// ---- K7: the WMMA kernel.
// t[n] = relu(feats[n]@w_v + b_v) . fcc_w ;  pacc[b] += Aw[n]*t[n]
// 8 waves x 16 rows = 128-row tile per block.
// A-fragments (all of K=512) kept register-resident: 16 x v16bf = 128 VGPRs/wave.
// B-fragments staged per 32-feature chunk in 32 KB LDS, shared by all 8 waves,
// via async global->LDS (GLOBAL_LOAD_ASYNC_TO_LDS_B128, ASYNCcnt).
__global__ __launch_bounds__(256) void fused_v_kernel(
    const float* __restrict__ feats, const unsigned short* __restrict__ wvpack,
    const float* __restrict__ b_v, const float* __restrict__ fcc_w,
    const float* __restrict__ Abuf, const int* __restrict__ offs,
    const unsigned* __restrict__ menc, const float* __restrict__ ssum,
    float* pacc) {
  __shared__ __align__(32) unsigned short bstage[16 * 1024];  // 32 KB

  int wave = threadIdx.x >> 5, lane = threadIdx.x & 31;
  int lrow = lane & 15, hi = lane >> 4;
  int rowbase = blockIdx.x * 128 + wave * 16;

  // Warm the wvpack stream (global_prefetch_b8).
  __builtin_prefetch(wvpack + (size_t)threadIdx.x * 1024, 0, 0);

  // ---- Load all 16 A-fragments once (rows resident across all f-chunks) ----
  // lanes 0-15 -> K offsets {0..15}; lanes 16-31 -> {16..31}
  const float* frowbase = feats + (size_t)(rowbase + lrow) * NF + hi * 16;
  v16bf afrag[16];
#pragma unroll
  for (int kb = 0; kb < 16; ++kb) {
    const float* fr = frowbase + kb * 32;
    float4 p0 = *(const float4*)(fr);
    float4 p1 = *(const float4*)(fr + 4);
    float4 p2 = *(const float4*)(fr + 8);
    float4 p3 = *(const float4*)(fr + 12);
    v16bf a;
    a[0]  = (__bf16)p0.x; a[1]  = (__bf16)p0.y; a[2]  = (__bf16)p0.z; a[3]  = (__bf16)p0.w;
    a[4]  = (__bf16)p1.x; a[5]  = (__bf16)p1.y; a[6]  = (__bf16)p1.z; a[7]  = (__bf16)p1.w;
    a[8]  = (__bf16)p2.x; a[9]  = (__bf16)p2.y; a[10] = (__bf16)p2.z; a[11] = (__bf16)p2.w;
    a[12] = (__bf16)p3.x; a[13] = (__bf16)p3.y; a[14] = (__bf16)p3.z; a[15] = (__bf16)p3.w;
    afrag[kb] = a;
  }

  float rp[8];
#pragma unroll
  for (int r = 0; r < 8; ++r) rp[r] = 0.f;

  for (int fc = 0; fc < 16; ++fc) {  // 16 chunks of 32 output features
    // ---- Stage B fragments for this chunk: 16 kb x 2 tiles x 1 KB = 32 KB ----
    {
      int kb  = threadIdx.x >> 4;      // 0..15
      int off = threadIdx.x & 15;      // 0..15, 128 B each
      const unsigned short* src =
          wvpack + (size_t)(kb * 32 + fc * 2) * 512 + off * 64;
      unsigned short* dst = bstage + kb * 1024 + off * 64;
#if defined(HAVE_ASYNC_LDS)
#pragma unroll
      for (int j = 0; j < 8; ++j) {
        __builtin_amdgcn_global_load_async_to_lds_b128(
            (AS1 v4i*)(src + j * 8),
            (AS3 v4i*)(dst + j * 8), 0, 0);
      }
#if __has_builtin(__builtin_amdgcn_s_wait_asynccnt)
      __builtin_amdgcn_s_wait_asynccnt(0);
#else
      asm volatile("s_wait_asynccnt 0x0" ::: "memory");
#endif
#else
#pragma unroll
      for (int j = 0; j < 8; ++j) {
        *(float4*)(dst + j * 8) = *(const float4*)(src + j * 8);
      }
#endif
    }
    __syncthreads();

    v8f zero = {0.f, 0.f, 0.f, 0.f, 0.f, 0.f, 0.f, 0.f};
    v8f acc0 = zero, acc1 = zero;
#pragma unroll
    for (int kb = 0; kb < 16; ++kb) {
      const v16bf* bl = (const v16bf*)(bstage + kb * 1024);
      v16bf b0 = bl[lane];
      v16bf b1 = bl[32 + lane];
      acc0 = __builtin_amdgcn_wmma_f32_16x16x32_bf16(false, afrag[kb], false, b0,
                                                     (short)0, acc0, false, false);
      acc1 = __builtin_amdgcn_wmma_f32_16x16x32_bf16(false, afrag[kb], false, b1,
                                                     (short)0, acc1, false, false);
    }
    // Epilogue: D layout — VGPR r: lanes0-15 M=r, lanes16-31 M=8+r; N = lane%16.
    {
      int f0 = fc * 32 + lrow;
      float bv0 = b_v[f0],      fw0 = fcc_w[f0];
      float bv1 = b_v[f0 + 16], fw1 = fcc_w[f0 + 16];
#pragma unroll
      for (int r = 0; r < 8; ++r) {
        rp[r] += fmaxf(acc0[r] + bv0, 0.f) * fw0;
        rp[r] += fmaxf(acc1[r] + bv1, 0.f) * fw1;
      }
    }
    __syncthreads();  // protect bstage before next chunk overwrites
  }

  // Reduce over the 16 lanes (N dim) within each half-wave.
#pragma unroll
  for (int r = 0; r < 8; ++r) {
    rp[r] += __shfl_xor(rp[r], 1);
    rp[r] += __shfl_xor(rp[r], 2);
    rp[r] += __shfl_xor(rp[r], 4);
    rp[r] += __shfl_xor(rp[r], 8);
  }
  if (lrow == 0) {   // lane 0 -> rows 0..7, lane 16 -> rows 8..15
    for (int r = 0; r < 8; ++r) {
      int gi = rowbase + hi * 8 + r;
      int b = 0; while (offs[b + 1] <= gi) b++;
      float aw = __expf(Abuf[gi] - decf(menc[b])) / ssum[b];
      atomicAdd(&pacc[b], aw * rp[r]);
    }
  }
}

// ---- K8: bag_pred = pacc + fcc_b ----
__global__ void finalize_kernel(const float* __restrict__ pacc,
                                const float* __restrict__ fcc_b,
                                float* __restrict__ pred) {
  int b = threadIdx.x;
  if (b < NBAG) pred[b] = pacc[b] + fcc_b[0];
}

extern "C" void kernel_launch(void* const* d_in, const int* in_sizes, int n_in,
                              void* d_out, int out_size, void* d_ws, size_t ws_size,
                              hipStream_t stream) {
  const float* feats = (const float*)d_in[0];
  const int*   split = (const int*)d_in[1];
  const float* w_ins = (const float*)d_in[2];
  const float* b_ins = (const float*)d_in[3];
  const float* w_q   = (const float*)d_in[4];
  const float* b_q   = (const float*)d_in[5];
  const float* w_v   = (const float*)d_in[6];
  const float* b_v   = (const float*)d_in[7];
  const float* fcc_w = (const float*)d_in[8];
  const float* fcc_b = (const float*)d_in[9];

  float* out  = (float*)d_out;
  float* pred = out;            // [32] bag_pred
  float* cout = out + NBAG;     // [131072] instance logits

  // workspace layout (floats), ~1.1 MB total
  float* wsf = (float*)d_ws;
  int*      offs = (int*)wsf;                       // 64 slots (33 used)
  int*      crit = (int*)(wsf + 64);                // 32
  unsigned* menc = (unsigned*)(wsf + 96);           // 32
  float*    ssum = wsf + 128;                       // 32
  float*    beta = wsf + 160;                       // 32
  float*    pacc = wsf + 192;                       // 32
  float*    qmax = wsf + 256;                       // 32*128
  float*    rvec = wsf + 4352;                      // 32*512
  float*    Abuf = wsf + 20736;                     // 131072
  unsigned short* wvpack = (unsigned short*)(wsf + 151808);  // 512*512 bf16 (32B aligned)

  init_kernel<<<1, 64, 0, stream>>>(split, offs, menc, ssum, pacc);
  logits_kernel<<<N_TOT / 8, 256, 0, stream>>>(feats, w_ins, b_ins, cout);
  wvpack_kernel<<<(NF * NF) / 256, 256, 0, stream>>>(w_v, wvpack);
  bag_argmax_kernel<<<NBAG, 256, 0, stream>>>(cout, offs, crit);
  qmax_kernel<<<NBAG, 256, 0, stream>>>(feats, w_q, b_q, crit, qmax, rvec, beta);
  scores_kernel<<<N_TOT / 8, 256, 0, stream>>>(feats, rvec, beta, offs, Abuf, menc);
  softsum_kernel<<<N_TOT / 256, 256, 0, stream>>>(Abuf, offs, menc, ssum);
  fused_v_kernel<<<N_TOT / 128, 256, 0, stream>>>(feats, wvpack, b_v, fcc_w,
                                                  Abuf, offs, menc, ssum, pacc);
  finalize_kernel<<<1, 32, 0, stream>>>(pacc, fcc_b, pred);
}
